// GraphSAGEPrimitive_41807211659463
// MI455X (gfx1250) — matrix operand order repaired and checked by reading, not compiled, not run on hardware
//
#include <hip/hip_runtime.h>
#include <hip/hip_bf16.h>

// ---------------- CDNA5 WMMA vector types ----------------
typedef __attribute__((ext_vector_type(16))) _Float16 v16h;
typedef __attribute__((ext_vector_type(8)))  _Float16 v8h;
typedef __attribute__((ext_vector_type(8)))  float    v8f;

#define N_NODES   100000
#define FEAT      128
#define ROW_TILES 6250        // 100000 / 16, exact
#define W_ELEMS   (FEAT*FEAT) // 16384

// ---------------- zero fill ----------------
__global__ void zero_f32(float* __restrict__ p, int n) {
    int i = blockIdx.x * blockDim.x + threadIdx.x;
    if (i < n) p[i] = 0.0f;
}

// ---------------- convert all four weight matrices f32 -> f16 ----------------
// layout in dst: [Wl0 | Wr0 | Wl1 | Wr1], each 16384 halfs
__global__ void cvt_weights(const float* __restrict__ wl0, const float* __restrict__ wr0,
                            const float* __restrict__ wl1, const float* __restrict__ wr1,
                            _Float16* __restrict__ dst) {
    int i = blockIdx.x * blockDim.x + threadIdx.x;
    if (i >= W_ELEMS) return;
    dst[i]             = (_Float16)wl0[i];
    dst[i + W_ELEMS]   = (_Float16)wr0[i];
    dst[i + 2*W_ELEMS] = (_Float16)wl1[i];
    dst[i + 3*W_ELEMS] = (_Float16)wr1[i];
}

// ---------------- degree count: 1 thread / edge ----------------
__global__ void count_deg(const long long* __restrict__ dst_idx,
                          float* __restrict__ cnt, int E) {
    int e = blockIdx.x * blockDim.x + threadIdx.x;
    if (e < E) atomicAdd(&cnt[(int)dst_idx[e]], 1.0f);
}

// ---------------- scatter-add features: 1 wave32 / edge ----------------
// lane L handles columns [4L, 4L+3]: one coalesced float4 read, 4 f32 atomics.
__global__ void scatter_add(const float* __restrict__ feat,
                            const long long* __restrict__ src_idx,
                            const long long* __restrict__ dst_idx,
                            float* __restrict__ sum, int E) {
    int w    = (blockIdx.x * blockDim.x + threadIdx.x) >> 5;
    int lane = threadIdx.x & 31;
    if (w >= E) return;
    int s = (int)src_idx[w];
    int d = (int)dst_idx[w];
    float4 v = ((const float4*)(feat + (size_t)s * FEAT))[lane];
    float* drow = sum + (size_t)d * FEAT + lane * 4;
    atomicAdd(drow + 0, v.x);
    atomicAdd(drow + 1, v.y);
    atomicAdd(drow + 2, v.z);
    atomicAdd(drow + 3, v.w);
}

// ---------------- fused mean + dual GEMM + bias + ReLU (WMMA) ----------------
// Block: 256 threads (8 waves). Block 'tile' computes rows [16*tile, 16*tile+15],
// wave w computes columns [16w, 16w+15]. out = relu(mean@Wl^T + bl + x@Wr^T + br).
// A fragments staged in LDS as f16 (shared by all 8 waves); B fragments read
// from f16 weights in global (L2-resident, row-major == B[k][n] layout, no transpose).
__global__ void __launch_bounds__(256)
sage_gemm(const float* __restrict__ xin,      // [N,128] layer input
          const float* __restrict__ sum,      // [N,128] scatter sums
          const float* __restrict__ cnt,      // [N]     degrees
          const _Float16* __restrict__ Wl16,  // [128,128] row-major (out,in)
          const _Float16* __restrict__ Wr16,  // [128,128]
          const float* __restrict__ bl,
          const float* __restrict__ br,
          float* __restrict__ out) {          // [N,128]
    __shared__ __align__(32) _Float16 sMean[16 * FEAT];
    __shared__ __align__(32) _Float16 sX   [16 * FEAT];

    const int tile = blockIdx.x;

    // Stage A tiles: 512 float4-groups, 2 per thread. mean = sum / max(cnt,1).
    #pragma unroll
    for (int i = threadIdx.x; i < 512; i += 256) {
        int row  = i >> 5;          // 0..15
        int c4   = i & 31;          // float4 column group
        int grow = tile * 16 + row;
        float inv = 1.0f / fmaxf(cnt[grow], 1.0f);
        float4 s4 = ((const float4*)sum)[grow * 32 + c4];
        float4 x4 = ((const float4*)xin)[grow * 32 + c4];
        int o = row * FEAT + c4 * 4;
        sMean[o + 0] = (_Float16)(s4.x * inv);
        sMean[o + 1] = (_Float16)(s4.y * inv);
        sMean[o + 2] = (_Float16)(s4.z * inv);
        sMean[o + 3] = (_Float16)(s4.w * inv);
        sX[o + 0] = (_Float16)x4.x;
        sX[o + 1] = (_Float16)x4.y;
        sX[o + 2] = (_Float16)x4.z;
        sX[o + 3] = (_Float16)x4.w;
    }
    __syncthreads();

    const int lane = threadIdx.x & 31;
    const int wave = threadIdx.x >> 5;
    const int m    = lane & 15;   // A row / B,C column within tile
    const int kh   = lane >> 4;   // K-half select (this code's contiguous k order)
    const int n    = wave * 16 + m;  // global output column

    const _Float16* Am = &sMean[m * FEAT];
    const _Float16* Ax = &sX   [m * FEAT];
    const _Float16* Bl = &Wl16[(size_t)n * FEAT];  // B[k][n] == Wl[n][k]
    const _Float16* Br = &Wr16[(size_t)n * FEAT];
    __builtin_prefetch(Bl, 0, 1);
    __builtin_prefetch(Br, 0, 1);

    v8f acc = {};

    // Mean path: K = 0..127 through Wl
    #pragma unroll
    for (int kc = 0; kc < FEAT; kc += 32) {
        // A (16x32 f16): lane holds 16 contiguous halfs at K = kc + kh*16
        v8h a0 = *(const v8h*)(Am + kc + kh * 16);
        v8h a1 = *(const v8h*)(Am + kc + kh * 16 + 8);
        v16h a = __builtin_shufflevector(a0, a1, 0,1,2,3,4,5,6,7,8,9,10,11,12,13,14,15);
        // B (32x16 f16): lane holds 16 contiguous halfs at K = kc + kh*16
        v16h b = *(const v16h*)(Bl + kc + kh * 16);
        acc = __builtin_amdgcn_wmma_f32_16x16x32_f16(false, a, false, b,
                                                     (short)0, acc, false, false);
    }
    // Self path: K = 0..127 through Wr
    #pragma unroll
    for (int kc = 0; kc < FEAT; kc += 32) {
        v8h a0 = *(const v8h*)(Ax + kc + kh * 16);
        v8h a1 = *(const v8h*)(Ax + kc + kh * 16 + 8);
        v16h a = __builtin_shufflevector(a0, a1, 0,1,2,3,4,5,6,7,8,9,10,11,12,13,14,15);
        v16h b = *(const v16h*)(Br + kc + kh * 16);
        acc = __builtin_amdgcn_wmma_f32_16x16x32_f16(false, a, false, b,
                                                     (short)0, acc, false, false);
    }

    // C/D layout: VGPR r holds row (r + 8*kh); bias + ReLU fused at store.
    const float bias = bl[n] + br[n];
    const int grow0 = tile * 16 + kh * 8;
    #pragma unroll
    for (int r = 0; r < 8; ++r) {
        float v = acc[r] + bias;
        out[(size_t)(grow0 + r) * FEAT + n] = v > 0.0f ? v : 0.0f;
    }
}

// ---------------- host-side orchestration ----------------
extern "C" void kernel_launch(void* const* d_in, const int* in_sizes, int n_in,
                              void* d_out, int out_size, void* d_ws, size_t ws_size,
                              hipStream_t stream) {
    const float*     x   = (const float*)d_in[0];
    const long long* ei  = (const long long*)d_in[1];   // int64 [2,E]
    const float*     Wl0 = (const float*)d_in[2];
    const float*     bl0 = (const float*)d_in[3];
    const float*     Wr0 = (const float*)d_in[4];
    const float*     br0 = (const float*)d_in[5];
    const float*     Wl1 = (const float*)d_in[6];
    const float*     bl1 = (const float*)d_in[7];
    const float*     Wr1 = (const float*)d_in[8];
    const float*     br1 = (const float*)d_in[9];
    float* out = (float*)d_out;

    const int E = in_sizes[1] / 2;
    const long long* src = ei;
    const long long* dst = ei + E;

    // Workspace layout (floats): h[N*128] | sum[N*128] | cnt[N] | f16 weights
    float* h      = (float*)d_ws;
    float* sumbuf = h + (size_t)N_NODES * FEAT;
    float* cntbuf = sumbuf + (size_t)N_NODES * FEAT;
    _Float16* w16 = (_Float16*)(cntbuf + N_NODES);   // 32B-aligned by construction
    _Float16* wl0h = w16;
    _Float16* wr0h = w16 + W_ELEMS;
    _Float16* wl1h = w16 + 2 * W_ELEMS;
    _Float16* wr1h = w16 + 3 * W_ELEMS;

    const int NF = N_NODES * FEAT;

    // One-time per launch: weight conversion
    cvt_weights<<<(W_ELEMS + 255) / 256, 256, 0, stream>>>(Wl0, Wr0, Wl1, Wr1, w16);

    // Zero sum + cnt (contiguous), then degrees (same for both layers)
    zero_f32<<<(NF + N_NODES + 255) / 256, 256, 0, stream>>>(sumbuf, NF + N_NODES);
    count_deg<<<(E + 255) / 256, 256, 0, stream>>>(dst, cntbuf, E);

    // ---- Layer 1 ----
    scatter_add<<<(E * 32 + 255) / 256, 256, 0, stream>>>(x, src, dst, sumbuf, E);
    sage_gemm<<<ROW_TILES, 256, 0, stream>>>(x, sumbuf, cntbuf, wl0h, wr0h, bl0, br0, h);

    // ---- Layer 2 ----
    zero_f32<<<(NF + 255) / 256, 256, 0, stream>>>(sumbuf, NF);
    scatter_add<<<(E * 32 + 255) / 256, 256, 0, stream>>>(h, src, dst, sumbuf, E);
    sage_gemm<<<ROW_TILES, 256, 0, stream>>>(h, sumbuf, cntbuf, wl1h, wr1h, bl1, br1, out);
}
